// GCN_29575144800296
// MI455X (gfx1250) — matrix operand order repaired and checked by reading, not compiled, not run on hardware
//
#include <hip/hip_runtime.h>
#include <hip/hip_bf16.h>

#define NN 100000     // nodes (multiple of 16 -> wave-uniform M-tile validity)
#define NE 1000000    // edges
#define FIN 128
#define FHID 256
#define FOUT 40
#define FOUTP 48      // FOUT padded to WMMA tile

typedef __bf16 bf16_t;
typedef bf16_t       v16bf __attribute__((ext_vector_type(16)));
typedef float        v8f   __attribute__((ext_vector_type(8)));
typedef unsigned int v4u   __attribute__((ext_vector_type(4)));

union BFrag { v4u q[2]; v16bf v; };

__device__ __forceinline__ unsigned short f2bf(float x) {
    union { float f; unsigned int u; } c; c.f = x;
    unsigned int u = c.u;
    return (unsigned short)((u + 0x7FFFu + ((u >> 16) & 1u)) >> 16);
}

__device__ __forceinline__ v8f wmma_bf16(const BFrag& a, const BFrag& b, v8f c) {
    return __builtin_amdgcn_wmma_f32_16x16x32_bf16(false, a.v, false, b.v,
                                                   (short)0, c, false, false);
}

// ---------------- degree / normalization ----------------

__global__ void k_fill1(float* __restrict__ deg, int n) {
    int i = blockIdx.x * blockDim.x + threadIdx.x;
    if (i < n) deg[i] = 1.0f;          // self-loop
}

__global__ void k_deg_count(float* __restrict__ deg, const int* __restrict__ dst, int e) {
    int i = blockIdx.x * blockDim.x + threadIdx.x;
    if (i < e) atomicAdd(&deg[dst[i]], 1.0f);
}

__global__ void k_rsqrt_inplace(float* __restrict__ d, int n) {
    int i = blockIdx.x * blockDim.x + threadIdx.x;
    if (i < n) {
        float v = d[i];
        d[i] = (v > 0.0f) ? rsqrtf(v) : 0.0f;
    }
}

// ---------------- fp32 -> bf16 streaming convert (optionally fused ReLU) ----------------

__global__ void k_f32_to_bf16(const float* __restrict__ in, unsigned short* __restrict__ outp,
                              int n4, int do_relu) {
    int i = blockIdx.x * blockDim.x + threadIdx.x;
    if (i < n4) {
        float4 v = ((const float4*)in)[i];
        if (do_relu) {
            v.x = fmaxf(v.x, 0.0f); v.y = fmaxf(v.y, 0.0f);
            v.z = fmaxf(v.z, 0.0f); v.w = fmaxf(v.w, 0.0f);
        }
        uint2 o;
        o.x = (unsigned)f2bf(v.x) | ((unsigned)f2bf(v.y) << 16);
        o.y = (unsigned)f2bf(v.z) | ((unsigned)f2bf(v.w) << 16);
        ((uint2*)outp)[i] = o;
    }
}

// W1 [FIN,FHID] fp32 -> W1t [FHID,FIN] bf16
__global__ void k_tr_w1(const float* __restrict__ W1, unsigned short* __restrict__ W1t) {
    int idx = blockIdx.x * blockDim.x + threadIdx.x;      // over FIN*FHID
    if (idx < FIN * FHID) {
        int k = idx / FHID, n = idx - k * FHID;
        W1t[n * FIN + k] = f2bf(W1[idx]);
    }
}

// W2 [FHID,FOUT] fp32 -> W2t [FOUTP,FHID] bf16, zero pad n>=FOUT
__global__ void k_tr_w2(const float* __restrict__ W2, unsigned short* __restrict__ W2t) {
    int idx = blockIdx.x * blockDim.x + threadIdx.x;      // over FOUTP*FHID
    if (idx < FOUTP * FHID) {
        int n = idx / FHID, k = idx - n * FHID;
        float v = (n < FOUT) ? W2[k * FOUT + n] : 0.0f;
        W2t[idx] = f2bf(v);
    }
}

// ---------------- GEMM1: H1[NN,256] = Xb[NN,128] @ W1t^T  (bf16 WMMA, no LDS) ----------------
// block = 8 waves: 4 along M (16 rows each) x 2 along N (32 cols each) => 64x64 tile

__global__ __launch_bounds__(256) void k_gemm1(const unsigned short* __restrict__ Xb,
                                               const unsigned short* __restrict__ W1t,
                                               float* __restrict__ H1) {
    const int tid  = threadIdx.x;
    const int lane = tid & 31;
    const int wave = tid >> 5;
    const int wm   = wave & 3;
    const int wn   = wave >> 2;
    const int rowBase = blockIdx.x * 64 + wm * 16;
    const int colBase = blockIdx.y * 64 + wn * 32;

    // NN % 16 == 0 -> whole 16-row wave tile valid or invalid (wave-uniform, EXEC stays all-1s)
    if (rowBase >= NN) return;

    // per-lane fragment gather bases (contiguous k = 16*(lane>>4) + s, same for A and B -> two 16B runs)
    const int m   = rowBase + (lane & 15);
    const int kbA = (lane >> 4) * 16;
    const int kbB = (lane >> 4) * 16;
    const int n0  = colBase + (lane & 15);

    const unsigned short* pa  = Xb  + (size_t)m * FIN + kbA;
    const unsigned short* pb0 = W1t + (size_t)n0 * FIN + kbB;
    const unsigned short* pb1 = W1t + (size_t)(n0 + 16) * FIN + kbB;

    v8f acc0 = {}; v8f acc1 = {};
    #pragma unroll
    for (int kt = 0; kt < FIN; kt += 32) {
        BFrag a, b0, b1;
        a.q[0]  = *(const v4u*)(pa + kt);          // k = kb..kb+7
        a.q[1]  = *(const v4u*)(pa + kt + 8);      // k = kb+8..kb+15
        b0.q[0] = *(const v4u*)(pb0 + kt);         // k = kb..kb+7
        b0.q[1] = *(const v4u*)(pb0 + kt + 8);     // k = kb+8..kb+15
        b1.q[0] = *(const v4u*)(pb1 + kt);
        b1.q[1] = *(const v4u*)(pb1 + kt + 8);
        acc0 = wmma_bf16(a, b0, acc0);
        acc1 = wmma_bf16(a, b1, acc1);
    }

    // C/D layout: VGPR r -> M = r + 8*(lane>>4), N = lane&15 ; no guards needed
    const int col0  = colBase + (lane & 15);
    const int rbase = rowBase + ((lane >> 4) * 8);
    #pragma unroll
    for (int r = 0; r < 8; ++r) {
        const size_t row = (size_t)(rbase + r);
        H1[row * FHID + col0]      = acc0[r];
        H1[row * FHID + col0 + 16] = acc1[r];
    }
}

// ---------------- layer-1 aggregation ----------------

__global__ __launch_bounds__(256) void k_agg1_init(const float* __restrict__ H1,
                                                   const float* __restrict__ dinv,
                                                   const float* __restrict__ b1,
                                                   float* __restrict__ A1) {
    int i = blockIdx.x;            // node
    int f = threadIdx.x;           // feature (256)
    float di = dinv[i];
    A1[(size_t)i * FHID + f] = b1[f] + H1[(size_t)i * FHID + f] * di * di;
}

__global__ __launch_bounds__(256) void k_agg1_edges(const float* __restrict__ H1,
                                                    const float* __restrict__ dinv,
                                                    const int* __restrict__ src,
                                                    const int* __restrict__ dst,
                                                    float* __restrict__ A1) {
    int e = blockIdx.x;            // edge (uniform -> scalar loads)
    int f = threadIdx.x;
    int s = src[e], d = dst[e];
    float norm = dinv[s] * dinv[d];
    atomicAdd(&A1[(size_t)d * FHID + f], H1[(size_t)s * FHID + f] * norm);
}

// ---------------- GEMM2: H2[NN,40] = A1b[NN,256] @ W2t^T (bf16 WMMA, no LDS) ----------------
// block = 8 waves along M (16 rows each) => 128 rows; each wave covers 48 cols (3 accums)

__global__ __launch_bounds__(256) void k_gemm2(const unsigned short* __restrict__ A1b,
                                               const unsigned short* __restrict__ W2t,
                                               float* __restrict__ H2) {
    const int tid  = threadIdx.x;
    const int lane = tid & 31;
    const int wave = tid >> 5;
    const int rowBase = blockIdx.x * 128 + wave * 16;

    if (rowBase >= NN) return;     // wave-uniform (NN % 16 == 0)

    const int m   = rowBase + (lane & 15);
    const int kbA = (lane >> 4) * 16;
    const int kbB = (lane >> 4) * 16;
    const int n0  = lane & 15;

    const unsigned short* pa  = A1b + (size_t)m * FHID + kbA;
    const unsigned short* pb0 = W2t + (size_t)n0 * FHID + kbB;
    const unsigned short* pb1 = W2t + (size_t)(n0 + 16) * FHID + kbB;
    const unsigned short* pb2 = W2t + (size_t)(n0 + 32) * FHID + kbB;

    v8f acc0 = {}; v8f acc1 = {}; v8f acc2 = {};
    #pragma unroll
    for (int kt = 0; kt < FHID; kt += 32) {
        BFrag a, b0, b1, b2;
        a.q[0]  = *(const v4u*)(pa + kt);
        a.q[1]  = *(const v4u*)(pa + kt + 8);
        b0.q[0] = *(const v4u*)(pb0 + kt);
        b0.q[1] = *(const v4u*)(pb0 + kt + 8);
        b1.q[0] = *(const v4u*)(pb1 + kt);
        b1.q[1] = *(const v4u*)(pb1 + kt + 8);
        b2.q[0] = *(const v4u*)(pb2 + kt);
        b2.q[1] = *(const v4u*)(pb2 + kt + 8);
        acc0 = wmma_bf16(a, b0, acc0);
        acc1 = wmma_bf16(a, b1, acc1);
        acc2 = wmma_bf16(a, b2, acc2);
    }

    const int rbase = rowBase + ((lane >> 4) * 8);
    const int c0 = lane & 15;
    #pragma unroll
    for (int r = 0; r < 8; ++r) {
        const size_t row = (size_t)(rbase + r);
        H2[row * FOUT + c0]      = acc0[r];
        H2[row * FOUT + c0 + 16] = acc1[r];
        if (c0 + 32 < FOUT)                      // only lanes with col < 40 (48-pad)
            H2[row * FOUT + c0 + 32] = acc2[r];
    }
}

// ---------------- layer-2 aggregation (into d_out) ----------------

__global__ void k_agg2_init(const float* __restrict__ H2,
                            const float* __restrict__ dinv,
                            const float* __restrict__ b2,
                            float* __restrict__ out, int total) {
    int idx = blockIdx.x * blockDim.x + threadIdx.x;
    if (idx < total) {
        int i = idx / FOUT;
        int f = idx - i * FOUT;
        float di = dinv[i];
        out[idx] = b2[f] + H2[idx] * di * di;
    }
}

__global__ void k_agg2_edges(const float* __restrict__ H2,
                             const float* __restrict__ dinv,
                             const int* __restrict__ src,
                             const int* __restrict__ dst,
                             float* __restrict__ out, int total) {
    int idx = blockIdx.x * blockDim.x + threadIdx.x;
    if (idx < total) {
        int e = idx / FOUT;
        int f = idx - e * FOUT;
        int s = src[e], d = dst[e];
        float norm = dinv[s] * dinv[d];
        atomicAdd(&out[(size_t)d * FOUT + f], H2[(size_t)s * FOUT + f] * norm);
    }
}

// ---------------- per-node log_softmax over 40 classes (in place) ----------------

__global__ void k_logsoftmax(float* __restrict__ out, int n) {
    int i = blockIdx.x * blockDim.x + threadIdx.x;
    if (i >= n) return;
    float* row = out + (size_t)i * FOUT;
    float mx = -3.4e38f;
    for (int c = 0; c < FOUT; ++c) mx = fmaxf(mx, row[c]);
    float s = 0.0f;
    for (int c = 0; c < FOUT; ++c) s += __expf(row[c] - mx);
    float ls = __logf(s);
    for (int c = 0; c < FOUT; ++c) row[c] = row[c] - mx - ls;
}

// ---------------- launch ----------------

extern "C" void kernel_launch(void* const* d_in, const int* in_sizes, int n_in,
                              void* d_out, int out_size, void* d_ws, size_t ws_size,
                              hipStream_t stream) {
    const float* x    = (const float*)d_in[0];
    const int*   edge = (const int*)  d_in[1];   // [2, NE] flat
    const float* W1   = (const float*)d_in[2];
    const float* b1   = (const float*)d_in[3];
    const float* W2   = (const float*)d_in[4];
    const float* b2   = (const float*)d_in[5];
    float* out = (float*)d_out;

    const int* src = edge;
    const int* dst = edge + NE;

    // workspace layout (16B-aligned offsets):
    //   [dinv 1MB][Xb bf16 25.6MB][W1t 64KB][W2t 24KB][H1 fp32 102.4MB][A1 fp32 102.4MB]
    //   A1b (bf16, 51.2MB) and H2 (fp32, 16MB) reuse H1's region once H1 is dead.
    char* ws = (char*)d_ws;
    size_t o = 0;
    float*          dinv = (float*)(ws + o);           o += (1u << 20);
    unsigned short* Xb   = (unsigned short*)(ws + o);  o += (size_t)NN * FIN * 2;     // 25.6MB
    unsigned short* W1t  = (unsigned short*)(ws + o);  o += (size_t)FHID * FIN * 2;   // 64KB
    unsigned short* W2t  = (unsigned short*)(ws + o);  o += (size_t)FOUTP * FHID * 2; // 24KB
    float*          H1   = (float*)(ws + o);           size_t oH1 = o; o += (size_t)NN * FHID * 4;
    float*          A1   = (float*)(ws + o);
    unsigned short* A1b  = (unsigned short*)(ws + oH1);                 // aliases dead H1
    float*          H2   = (float*)(ws + oH1 + (size_t)NN * FHID * 2);  // after A1b

    // symmetric degree normalization (self-loops included)
    k_fill1<<<(NN + 255) / 256, 256, 0, stream>>>(dinv, NN);
    k_deg_count<<<(NE + 255) / 256, 256, 0, stream>>>(dinv, dst, NE);
    k_rsqrt_inplace<<<(NN + 255) / 256, 256, 0, stream>>>(dinv, NN);

    // operand preparation: bf16 copies laid out for direct WMMA fragment loads
    int n4x = NN * FIN / 4;
    k_f32_to_bf16<<<(n4x + 255) / 256, 256, 0, stream>>>(x, Xb, n4x, 0);
    k_tr_w1<<<(FIN * FHID + 255) / 256, 256, 0, stream>>>(W1, W1t);
    k_tr_w2<<<(FOUTP * FHID + 255) / 256, 256, 0, stream>>>(W2, W2t);

    // layer 1
    dim3 g1((NN + 63) / 64, FHID / 64);
    k_gemm1<<<g1, 256, 0, stream>>>(Xb, W1t, H1);
    k_agg1_init<<<NN, FHID, 0, stream>>>(H1, dinv, b1, A1);
    k_agg1_edges<<<NE, FHID, 0, stream>>>(H1, dinv, src, dst, A1);

    // layer 2: ReLU fused into the bf16 convert; then no-LDS WMMA GEMM
    int n4a = NN * FHID / 4;
    k_f32_to_bf16<<<(n4a + 255) / 256, 256, 0, stream>>>(A1, A1b, n4a, 1);
    k_gemm2<<<(NN + 127) / 128, 256, 0, stream>>>(A1b, W2t, H2);

    int tot2 = NN * FOUT;
    k_agg2_init<<<(tot2 + 255) / 256, 256, 0, stream>>>(H2, dinv, b2, out, tot2);
    int tote = NE * FOUT;   // 40,000,000
    k_agg2_edges<<<(tote + 255) / 256, 256, 0, stream>>>(H2, dinv, src, dst, out, tote);

    k_logsoftmax<<<(NN + 255) / 256, 256, 0, stream>>>(out, NN);
}
